// BertTagger_13469017440311
// MI455X (gfx1250) — matrix operand order repaired and checked by reading, not compiled, not run on hardware
//
#include <hip/hip_runtime.h>
#include <hip/hip_bf16.h>
#include <math.h>

// ---------------------------------------------------------------------------
// Problem constants (reference: BERT tagger, eval mode)
// ---------------------------------------------------------------------------
#define BB   12
#define LL   512
#define DD   768
#define HH   12
#define DH   64
#define NLAY 4
#define FF_  3072
#define NLAB 4
#define NREL 5
#define NQ   3

typedef __attribute__((ext_vector_type(16))) _Float16 v16h;
typedef __attribute__((ext_vector_type(8)))  _Float16 v8h;
typedef __attribute__((ext_vector_type(8)))  float    v8f;

__device__ inline float gelu_tanh(float x) {
    float x3 = x * x * x;
    return 0.5f * x * (1.0f + tanhf(0.7978845608028654f * (x + 0.044715f * x3)));
}

// ---------------------------------------------------------------------------
// WMMA GEMM: C = act(alpha * A(f16) @ B(f16) + bias)
//   block tile 128x128, K tile 64, 256 threads = 8 waves (wave32)
//   each wave computes 32x64 via 2x4 v_wmma_f32_16x16x32_f16 tiles.
//   Fast path (full tiles, K % 64 == 0): vectorized b128 staging with
//   register-based double buffering (global loads issued before the WMMA
//   block, spilled to the alternate LDS buffer after).
//   Batched via blockIdx.z with per-matrix strides (per-head attention).
//   transB: B stored [N,K] row-major (used for Q @ K^T).
//   act: 0 = none, 1 = gelu(tanh approx)
// ---------------------------------------------------------------------------
#define TM 128
#define TN 128
#define TK 64

__global__ __launch_bounds__(256)
void wmma_gemm_kernel(const _Float16* __restrict__ A,
                      const _Float16* __restrict__ B,
                      const float*    __restrict__ bias,
                      float*          __restrict__ Cf,
                      _Float16*       __restrict__ Ch,
                      int M, int N, int K,
                      int lda, int ldb, int ldc,
                      long long strideA, long long strideB, long long strideC,
                      int transB, int act, float alpha)
{
    __shared__ _Float16 As[2][TM][TK];   // row-major tiles of A
    __shared__ _Float16 Bs[2][TN][TK];   // K-transposed tiles of B: Bs[n][k]

    const int tid  = threadIdx.x;
    const int lane = tid & 31;
    const int wave = tid >> 5;
    const int wm   = wave & 3;    // 4 waves along M (4*32 = 128)
    const int wn   = wave >> 2;   // 2 waves along N (2*64 = 128)
    const int bm   = blockIdx.y * TM;
    const int bn   = blockIdx.x * TN;
    const int z    = blockIdx.z;

    A += (long long)z * strideA;
    B += (long long)z * strideB;
    const long long co = (long long)z * strideC;

    const int halfLo = (lane < 16) ? 1 : 0;
    const int l16    = lane & 15;

    v8f acc[2][4];
#pragma unroll
    for (int mi = 0; mi < 2; ++mi)
#pragma unroll
        for (int ni = 0; ni < 4; ++ni)
#pragma unroll
            for (int j = 0; j < 8; ++j) acc[mi][ni][j] = 0.0f;

    // ---- fragment compute from LDS buffer `buf` ----
    auto compute = [&](int buf) {
        _Float16 (*Asb)[TK] = As[buf];
        _Float16 (*Bsb)[TK] = Bs[buf];
#pragma unroll
        for (int ks = 0; ks < TK; ks += 32) {
            v16h afrag[2];
#pragma unroll
            for (int mi = 0; mi < 2; ++mi) {
                int row = wm * 32 + mi * 16 + l16;
                unsigned int* au = reinterpret_cast<unsigned int*>(&afrag[mi]);
                int ka = ks + (halfLo ? 0 : 16);
                int kb = ks + (halfLo ? 8 : 24);
#pragma unroll
                for (int r = 0; r < 4; ++r)
                    au[r] = *(const unsigned int*)&Asb[row][ka + 2 * r];
#pragma unroll
                for (int r = 0; r < 4; ++r)
                    au[4 + r] = *(const unsigned int*)&Asb[row][kb + 2 * r];
            }
            v16h bfrag[4];
#pragma unroll
            for (int ni = 0; ni < 4; ++ni) {
                int col = wn * 64 + ni * 16 + l16;
                unsigned int* bu = reinterpret_cast<unsigned int*>(&bfrag[ni]);
                int kb = ks + (halfLo ? 0 : 16);
#pragma unroll
                for (int r = 0; r < 8; ++r)
                    bu[r] = *(const unsigned int*)&Bsb[col][kb + 2 * r];
            }
#pragma unroll
            for (int mi = 0; mi < 2; ++mi)
#pragma unroll
                for (int ni = 0; ni < 4; ++ni)
                    acc[mi][ni] = __builtin_amdgcn_wmma_f32_16x16x32_f16(
                        false, afrag[mi], false, bfrag[ni],
                        (short)0, acc[mi][ni], false, false);
        }
    };

    const bool fast = (bm + TM <= M) && (bn + TN <= N) && ((K & (TK - 1)) == 0);

    if (fast) {
        // per-thread b128 chunk indices
        // A / transB-B: li in [0,1024): row = li>>3, kcol = (li&7)*8
        // non-trans B : li in [0,1024): krow = li>>4, ncol = (li&15)*8
        v8h areg[4], breg[4];
        auto loadTile = [&](int k0) {
#pragma unroll
            for (int it = 0; it < 4; ++it) {
                int li = tid + 256 * it;
                areg[it] = *(const v8h*)&A[(long long)(bm + (li >> 3)) * lda + k0 + ((li & 7) << 3)];
            }
            if (transB) {
#pragma unroll
                for (int it = 0; it < 4; ++it) {
                    int li = tid + 256 * it;
                    breg[it] = *(const v8h*)&B[(long long)(bn + (li >> 3)) * ldb + k0 + ((li & 7) << 3)];
                }
            } else {
#pragma unroll
                for (int it = 0; it < 4; ++it) {
                    int li = tid + 256 * it;
                    breg[it] = *(const v8h*)&B[(long long)(k0 + (li >> 4)) * ldb + bn + ((li & 15) << 3)];
                }
            }
        };
        auto storeTile = [&](int buf) {
#pragma unroll
            for (int it = 0; it < 4; ++it) {
                int li = tid + 256 * it;
                *(v8h*)&As[buf][li >> 3][(li & 7) << 3] = areg[it];
            }
            if (transB) {
#pragma unroll
                for (int it = 0; it < 4; ++it) {
                    int li = tid + 256 * it;
                    *(v8h*)&Bs[buf][li >> 3][(li & 7) << 3] = breg[it];
                }
            } else {
#pragma unroll
                for (int it = 0; it < 4; ++it) {
                    int li = tid + 256 * it;
                    int kk = li >> 4, n0 = (li & 15) << 3;
#pragma unroll
                    for (int e = 0; e < 8; ++e) Bs[buf][n0 + e][kk] = breg[it][e];
                }
            }
        };

        const int nk = K / TK;
        loadTile(0);
        storeTile(0);
        int cur = 0;
        for (int t = 0; t < nk; ++t) {
            __syncthreads();                       // staged tile t visible
            if (t + 1 < nk) loadTile((t + 1) * TK); // issue next-tile global loads
            compute(cur);                           // WMMA from current buffer
            if (t + 1 < nk) storeTile(1 - cur);     // spill regs -> other buffer
            cur ^= 1;
        }
    } else {
        // generic guarded path (single buffer)
        for (int k0 = 0; k0 < K; k0 += TK) {
#pragma unroll 4
            for (int i = 0; i < 32; ++i) {
                int li = tid + 256 * i;
                int kk = li & (TK - 1);
                int rr = li >> 6;
                int gr = bm + rr, gk = k0 + kk;
                _Float16 v = (_Float16)0.0f;
                if (gr < M && gk < K) v = A[(long long)gr * lda + gk];
                As[0][rr][kk] = v;
            }
            if (transB) {
#pragma unroll 4
                for (int i = 0; i < 32; ++i) {
                    int li = tid + 256 * i;
                    int kk = li & (TK - 1);
                    int nn = li >> 6;
                    int gn = bn + nn, gk = k0 + kk;
                    _Float16 v = (_Float16)0.0f;
                    if (gn < N && gk < K) v = B[(long long)gn * ldb + gk];
                    Bs[0][nn][kk] = v;
                }
            } else {
#pragma unroll 4
                for (int i = 0; i < 32; ++i) {
                    int li = tid + 256 * i;
                    int nn = li & (TN - 1);
                    int kk = li >> 7;
                    int gn = bn + nn, gk = k0 + kk;
                    _Float16 v = (_Float16)0.0f;
                    if (gn < N && gk < K) v = B[(long long)gk * ldb + gn];
                    Bs[0][nn][kk] = v;
                }
            }
            __syncthreads();
            compute(0);
            __syncthreads();
        }
    }

    // --- epilogue: C/D layout: VGPR j -> M = j + (lane<16?0:8), N = lane&15 ---
#pragma unroll
    for (int mi = 0; mi < 2; ++mi)
#pragma unroll
        for (int ni = 0; ni < 4; ++ni)
#pragma unroll
            for (int j = 0; j < 8; ++j) {
                int row = bm + wm * 32 + mi * 16 + j + (halfLo ? 0 : 8);
                int col = bn + wn * 64 + ni * 16 + l16;
                if (row < M && col < N) {
                    float v = acc[mi][ni][j] * alpha;
                    if (bias) v += bias[col];
                    if (act == 1) v = gelu_tanh(v);
                    long long o = co + (long long)row * ldc + col;
                    if (Cf) Cf[o] = v;
                    if (Ch) Ch[o] = (_Float16)v;
                }
            }
}

// ---------------------------------------------------------------------------
// f32 -> f16 conversion
// ---------------------------------------------------------------------------
__global__ void f32_to_f16_kernel(const float* __restrict__ s, _Float16* __restrict__ d, int n)
{
    int i = blockIdx.x * 256 + threadIdx.x;
    if (i < n) d[i] = (_Float16)s[i];
}

// ---------------------------------------------------------------------------
// Embedding gather + LayerNorm (one block per token row of 768)
// ---------------------------------------------------------------------------
__global__ __launch_bounds__(256)
void embed_ln_kernel(const int* __restrict__ ids, const int* __restrict__ tt,
                     const float* __restrict__ we, const float* __restrict__ pe,
                     const float* __restrict__ te,
                     const float* __restrict__ g, const float* __restrict__ b,
                     float* __restrict__ xo, _Float16* __restrict__ xho)
{
    __shared__ float red[256];
    int row = blockIdx.x;            // b*L + l
    int l   = row & (LL - 1);
    int t   = threadIdx.x;
    int id  = ids[row];
    int typ = tt[row];
    long long base = (long long)row * DD;
    float v[3];
#pragma unroll
    for (int j = 0; j < 3; ++j) {
        int i = t + 256 * j;
        v[j] = we[(long long)id * DD + i] + pe[(long long)l * DD + i] + te[(long long)typ * DD + i];
    }
    red[t] = v[0] + v[1] + v[2];
    __syncthreads();
    for (int o = 128; o > 0; o >>= 1) { if (t < o) red[t] += red[t + o]; __syncthreads(); }
    float mu = red[0] * (1.0f / DD);
    __syncthreads();
    float d2 = 0.f;
#pragma unroll
    for (int j = 0; j < 3; ++j) { float d = v[j] - mu; d2 += d * d; }
    red[t] = d2;
    __syncthreads();
    for (int o = 128; o > 0; o >>= 1) { if (t < o) red[t] += red[t + o]; __syncthreads(); }
    float rs = rsqrtf(red[0] * (1.0f / DD) + 1e-12f);
#pragma unroll
    for (int j = 0; j < 3; ++j) {
        int i = t + 256 * j;
        float o = (v[j] - mu) * rs * g[i] + b[i];
        xo[base + i] = o;
        xho[base + i] = (_Float16)o;
    }
}

// ---------------------------------------------------------------------------
// Residual add + LayerNorm: x = LN(x + y); writes f32 and f16 copies
// ---------------------------------------------------------------------------
__global__ __launch_bounds__(256)
void residual_ln_kernel(const float* __restrict__ x, const float* __restrict__ y,
                        const float* __restrict__ g, const float* __restrict__ b,
                        float* __restrict__ xo, _Float16* __restrict__ xho)
{
    __shared__ float red[256];
    int row = blockIdx.x;
    int t   = threadIdx.x;
    long long base = (long long)row * DD;
    float v[3];
#pragma unroll
    for (int j = 0; j < 3; ++j) {
        int i = t + 256 * j;
        v[j] = x[base + i] + y[base + i];
    }
    red[t] = v[0] + v[1] + v[2];
    __syncthreads();
    for (int o = 128; o > 0; o >>= 1) { if (t < o) red[t] += red[t + o]; __syncthreads(); }
    float mu = red[0] * (1.0f / DD);
    __syncthreads();
    float d2 = 0.f;
#pragma unroll
    for (int j = 0; j < 3; ++j) { float d = v[j] - mu; d2 += d * d; }
    red[t] = d2;
    __syncthreads();
    for (int o = 128; o > 0; o >>= 1) { if (t < o) red[t] += red[t + o]; __syncthreads(); }
    float rs = rsqrtf(red[0] * (1.0f / DD) + 1e-12f);
#pragma unroll
    for (int j = 0; j < 3; ++j) {
        int i = t + 256 * j;
        float o = (v[j] - mu) * rs * g[i] + b[i];
        xo[base + i] = o;
        xho[base + i] = (_Float16)o;
    }
}

// ---------------------------------------------------------------------------
// Softmax over 512 columns with attention-mask bias, emits f16 probabilities.
// scores layout: [H, L, L] for one batch b. grid = H*L blocks of 256.
// ---------------------------------------------------------------------------
__global__ __launch_bounds__(256)
void softmax_kernel(const float* __restrict__ scores, const int* __restrict__ mask,
                    _Float16* __restrict__ attnh, int b)
{
    __shared__ float red[256];
    int h   = blockIdx.x >> 9;
    int row = blockIdx.x & (LL - 1);
    int t   = threadIdx.x;
    const float* s = scores + ((long long)h * LL + row) * LL;
    _Float16*    o = attnh  + ((long long)h * LL + row) * LL;
    const int* mrow = mask + b * LL;
    float bias0 = (1.0f - (float)mrow[t])        * -10000.0f;
    float bias1 = (1.0f - (float)mrow[t + 256])  * -10000.0f;
    float v0 = s[t] + bias0;
    float v1 = s[t + 256] + bias1;
    red[t] = fmaxf(v0, v1);
    __syncthreads();
    for (int of = 128; of > 0; of >>= 1) { if (t < of) red[t] = fmaxf(red[t], red[t + of]); __syncthreads(); }
    float m = red[0];
    __syncthreads();
    float e0 = __expf(v0 - m), e1 = __expf(v1 - m);
    red[t] = e0 + e1;
    __syncthreads();
    for (int of = 128; of > 0; of >>= 1) { if (t < of) red[t] += red[t + of]; __syncthreads(); }
    float inv = 1.0f / red[0];
    o[t]       = (_Float16)(e0 * inv);
    o[t + 256] = (_Float16)(e1 * inv);
}

// ---------------------------------------------------------------------------
// Mean pool over L per (b, d)
// ---------------------------------------------------------------------------
__global__ void pool_kernel(const float* __restrict__ x, float* __restrict__ pool)
{
    int i = blockIdx.x * 256 + threadIdx.x;
    if (i >= BB * DD) return;
    int b = i / DD, d = i % DD;
    float s = 0.f;
    for (int l = 0; l < LL; ++l) s += x[((long long)(b * LL + l)) * DD + d];
    pool[i] = s * (1.0f / LL);
}

// ---------------------------------------------------------------------------
// Token classifier: logits[b,l,c] = x_row . cls_w[:,c] + cls_b[c]
// ---------------------------------------------------------------------------
__global__ void cls_kernel(const float* __restrict__ x, const float* __restrict__ w,
                           const float* __restrict__ bias, float* __restrict__ out)
{
    int i = blockIdx.x * 256 + threadIdx.x;
    if (i >= BB * LL * NLAB) return;
    int c = i & (NLAB - 1);
    int row = i >> 2;
    float s = bias[c];
    long long base = (long long)row * DD;
    for (int k = 0; k < DD; ++k) s += x[base + k] * w[k * NLAB + c];
    out[i] = s;
}

// serial prefix positions for valid-token compaction (1 thread per batch row)
__global__ void pos_kernel(const int* __restrict__ valid, int* __restrict__ pos)
{
    int b = blockIdx.x;
    int c = 0;
    for (int l = 0; l < LL; ++l) {
        int i = b * LL + l;
        if (valid[i] == 1) pos[i] = c++;
        else pos[i] = LL;
    }
}

__global__ void init_logits_kernel(const float* __restrict__ cls_b, float* __restrict__ lv)
{
    int i = blockIdx.x * 256 + threadIdx.x;
    if (i >= BB * LL) return;
#pragma unroll
    for (int c = 0; c < NLAB; ++c) lv[i * NLAB + c] = cls_b[c];
}

__global__ void scatter_kernel(const int* __restrict__ valid, const int* __restrict__ pos,
                               const float* __restrict__ tok, float* __restrict__ lv)
{
    int i = blockIdx.x * 256 + threadIdx.x;
    if (i >= BB * LL) return;
    if (valid[i] != 1) return;
    int b = i / LL;
    int p = pos[i];
    if (p >= LL) return;
    long long dst = ((long long)(b * LL + p)) * NLAB;
#pragma unroll
    for (int c = 0; c < NLAB; ++c) lv[dst + c] = tok[(long long)i * NLAB + c];
}

__global__ void logp_kernel(const float* __restrict__ lv, float* __restrict__ logp)
{
    int i = blockIdx.x * 256 + threadIdx.x;
    if (i >= BB * LL) return;
    float v[NLAB];
    float m = -1e30f;
#pragma unroll
    for (int c = 0; c < NLAB; ++c) { v[c] = lv[i * NLAB + c]; m = fmaxf(m, v[c]); }
    float s = 0.f;
#pragma unroll
    for (int c = 0; c < NLAB; ++c) s += __expf(v[c] - m);
    float lse = logf(s) + m;
#pragma unroll
    for (int c = 0; c < NLAB; ++c) logp[i * NLAB + c] = v[c] - lse;
}

// deterministic single-block masked-CE reduction
__global__ __launch_bounds__(256)
void ce_kernel(const float* __restrict__ logp, const int* __restrict__ labels,
               const int* __restrict__ aml, float* __restrict__ acc)
{
    __shared__ float sn[256], sm[256];
    int t = threadIdx.x;
    float a = 0.f, mt = 0.f;
    for (int i = t; i < BB * LL; i += 256) {
        int lab = labels[i];
        if (aml[i] == 1 && lab != 0) {
            a += -logp[(long long)i * NLAB + lab];
            mt += 1.0f;
        }
    }
    sn[t] = a; sm[t] = mt;
    __syncthreads();
    for (int o = 128; o > 0; o >>= 1) {
        if (t < o) { sn[t] += sn[t + o]; sm[t] += sm[t + o]; }
        __syncthreads();
    }
    if (t == 0) { acc[0] = sn[0]; acc[1] = sm[0]; }
}

// rel head raw logits: rel_raw[b*5+j] = pool[b] . rel_w[:,j] + rel_b[j]
__global__ void rel_kernel(const float* __restrict__ pool, const float* __restrict__ w,
                           const float* __restrict__ bias, float* __restrict__ rel_raw)
{
    int i = threadIdx.x;
    if (i >= BB * NREL) return;
    int b = i / NREL, j = i % NREL;
    float s = bias[j];
    for (int k = 0; k < DD; ++k) s += pool[b * DD + k] * w[k * NREL + j];
    rel_raw[i] = s;
}

// sigmoid + BCE + final loss combine, writes out[0] and rel_logits
__global__ void final_kernel(const float* __restrict__ rel_raw, const float* __restrict__ rel_labels,
                             const int* __restrict__ type_flag, const float* __restrict__ ce_acc,
                             float* __restrict__ out)
{
    float loss_ce = ce_acc[0] / fmaxf(ce_acc[1], 1.0f);
    float bce = 0.f, n_act = 0.f;
    float rl[BB / NQ][NREL];
    for (int g = 0; g < BB / NQ; ++g) {
        for (int j = 0; j < NREL; ++j) {
            float m = (rel_raw[(g * NQ + 0) * NREL + j] +
                       rel_raw[(g * NQ + 1) * NREL + j] +
                       rel_raw[(g * NQ + 2) * NREL + j]) * (1.0f / NQ);
            float sg = 1.0f / (1.0f + __expf(-m));
            rl[g][j] = sg;
            out[1 + BB * LL * NLAB + g * NREL + j] = sg;
        }
        if (type_flag[g] == 1) n_act += 1.0f;
    }
    for (int g = 0; g < BB / NQ; ++g) {
        float mf = (type_flag[g] == 1) ? 1.0f : 0.0f;
        for (int j = 0; j < NREL; ++j) {
            float pr = fminf(fmaxf(rl[g][j], 1e-12f), 1.0f - 1e-12f);
            float y  = rel_labels[g * NREL + j];
            bce += mf * (-(y * logf(pr) + (1.0f - y) * logf(1.0f - pr)));
        }
    }
    float loss_bce = bce / fmaxf(n_act * (float)NREL, 1.0f);
    out[0] = (n_act > 0.f) ? (loss_ce + loss_bce) : loss_ce;
}

// ---------------------------------------------------------------------------
// Host-side orchestration
// ---------------------------------------------------------------------------
extern "C" void kernel_launch(void* const* d_in, const int* in_sizes, int n_in,
                              void* d_out, int out_size, void* d_ws, size_t ws_size,
                              hipStream_t stream)
{
    (void)in_sizes; (void)n_in; (void)out_size; (void)ws_size;

    // ---- inputs: setup_inputs() dict order, params flattened by sorted keys ----
    const int*   input_ids  = (const int*)d_in[0];
    const int*   tok_type   = (const int*)d_in[1];
    const int*   attn_mask  = (const int*)d_in[2];
    const int*   labels     = (const int*)d_in[3];
    const int*   valid_ids  = (const int*)d_in[4];
    const int*   aml        = (const int*)d_in[5];
    const int*   type_flag  = (const int*)d_in[6];
    const float* rel_labels = (const float*)d_in[7];
    // params sorted: cls_b, cls_w, emb_ln_b, emb_ln_g, layers[0..3], pos_emb, rel_b, rel_w, type_emb, word_emb
    const float* cls_b    = (const float*)d_in[8];
    const float* cls_w    = (const float*)d_in[9];
    const float* emb_ln_b = (const float*)d_in[10];
    const float* emb_ln_g = (const float*)d_in[11];
    // per-layer sorted: b1,b2,bk,bo,bq,bv,ln1_b,ln1_g,ln2_b,ln2_g,w1,w2,wk,wo,wq,wv
    const float* Lp[NLAY][16];
    for (int l = 0; l < NLAY; ++l)
        for (int j = 0; j < 16; ++j)
            Lp[l][j] = (const float*)d_in[12 + l * 16 + j];
    enum { P_B1=0, P_B2, P_BK, P_BO, P_BQ, P_BV, P_LN1B, P_LN1G, P_LN2B, P_LN2G,
           P_W1, P_W2, P_WK, P_WO, P_WQ, P_WV };
    const float* pos_emb  = (const float*)d_in[12 + NLAY * 16 + 0];
    const float* rel_b    = (const float*)d_in[12 + NLAY * 16 + 1];
    const float* rel_w    = (const float*)d_in[12 + NLAY * 16 + 2];
    const float* type_emb = (const float*)d_in[12 + NLAY * 16 + 3];
    const float* word_emb = (const float*)d_in[12 + NLAY * 16 + 4];

    float* out = (float*)d_out;

    // ---- workspace bump allocator ----
    char*  wsp = (char*)d_ws;
    size_t off = 0;
    auto alloc = [&](size_t bytes) -> void* {
        void* p = wsp + off;
        off += (bytes + 255) & ~(size_t)255;
        return p;
    };
    const long long S  = (long long)BB * LL * DD;   // 4.7M elems
    const long long FB = (long long)BB * LL * FF_;  // 18.9M elems
    const long long SS = (long long)HH * LL * LL;   // per-batch scores

    float*    x      = (float*)   alloc(S * 4);
    float*    y      = (float*)   alloc(S * 4);
    _Float16* xh     = (_Float16*)alloc(S * 2);
    _Float16* qh     = (_Float16*)alloc(S * 2);
    _Float16* kh     = (_Float16*)alloc(S * 2);
    _Float16* vh     = (_Float16*)alloc(S * 2);
    _Float16* ctxh   = (_Float16*)alloc(S * 2);
    _Float16* hh     = (_Float16*)alloc(FB * 2);
    float*    scores = (float*)   alloc(SS * 4);
    _Float16* attnh  = (_Float16*)alloc(SS * 2);
    _Float16* wq_h[NLAY]; _Float16* wk_h[NLAY]; _Float16* wv_h[NLAY];
    _Float16* wo_h[NLAY]; _Float16* w1_h[NLAY]; _Float16* w2_h[NLAY];
    for (int l = 0; l < NLAY; ++l) {
        wq_h[l] = (_Float16*)alloc((size_t)DD * DD * 2);
        wk_h[l] = (_Float16*)alloc((size_t)DD * DD * 2);
        wv_h[l] = (_Float16*)alloc((size_t)DD * DD * 2);
        wo_h[l] = (_Float16*)alloc((size_t)DD * DD * 2);
        w1_h[l] = (_Float16*)alloc((size_t)DD * FF_ * 2);
        w2_h[l] = (_Float16*)alloc((size_t)FF_ * DD * 2);
    }
    float* pool     = (float*)alloc((size_t)BB * DD * 4);
    float* tok_log  = (float*)alloc((size_t)BB * LL * NLAB * 4);
    float* logits_v = (float*)alloc((size_t)BB * LL * NLAB * 4);
    int*   posbuf   = (int*)  alloc((size_t)BB * LL * 4);
    float* rel_raw  = (float*)alloc((size_t)BB * NREL * 4);
    float* ce_acc   = (float*)alloc(8 * 4);

    auto conv = [&](const float* s, _Float16* d, long long n) {
        f32_to_f16_kernel<<<dim3((unsigned)((n + 255) / 256)), dim3(256), 0, stream>>>(s, d, (int)n);
    };
    auto gemm = [&](const _Float16* A, const _Float16* Bm, const float* bias,
                    float* Cf, _Float16* Ch, int M, int N, int K,
                    int lda, int ldb, int ldc,
                    long long sA, long long sB, long long sC,
                    int batch, int transB, int act, float alpha) {
        dim3 grid((N + TN - 1) / TN, (M + TM - 1) / TM, batch);
        wmma_gemm_kernel<<<grid, dim3(256), 0, stream>>>(A, Bm, bias, Cf, Ch,
            M, N, K, lda, ldb, ldc, sA, sB, sC, transB, act, alpha);
    };

    // ---- weight conversion (once per launch; deterministic) ----
    for (int l = 0; l < NLAY; ++l) {
        conv(Lp[l][P_WQ], wq_h[l], (long long)DD * DD);
        conv(Lp[l][P_WK], wk_h[l], (long long)DD * DD);
        conv(Lp[l][P_WV], wv_h[l], (long long)DD * DD);
        conv(Lp[l][P_WO], wo_h[l], (long long)DD * DD);
        conv(Lp[l][P_W1], w1_h[l], (long long)DD * FF_);
        conv(Lp[l][P_W2], w2_h[l], (long long)FF_ * DD);
    }

    // ---- embeddings + LN ----
    embed_ln_kernel<<<dim3(BB * LL), dim3(256), 0, stream>>>(
        input_ids, tok_type, word_emb, pos_emb, type_emb, emb_ln_g, emb_ln_b, x, xh);

    const int ML = BB * LL;  // 6144
    // ---- encoder layers ----
    for (int l = 0; l < NLAY; ++l) {
        // QKV projections (f16 outputs)
        gemm(xh, wq_h[l], Lp[l][P_BQ], nullptr, qh, ML, DD, DD, DD, DD, DD, 0, 0, 0, 1, 0, 0, 1.0f);
        gemm(xh, wk_h[l], Lp[l][P_BK], nullptr, kh, ML, DD, DD, DD, DD, DD, 0, 0, 0, 1, 0, 0, 1.0f);
        gemm(xh, wv_h[l], Lp[l][P_BV], nullptr, vh, ML, DD, DD, DD, DD, DD, 0, 0, 0, 1, 0, 0, 1.0f);
        // attention, looped over batch, batched over heads via blockIdx.z
        for (int b = 0; b < BB; ++b) {
            const _Float16* qb = qh + (long long)b * LL * DD;
            const _Float16* kb = kh + (long long)b * LL * DD;
            const _Float16* vb = vh + (long long)b * LL * DD;
            // scores[h] = (1/8) * Q_h @ K_h^T   (per-head stride 64 in D)
            gemm(qb, kb, nullptr, scores, nullptr, LL, LL, DH,
                 DD, DD, LL, DH, DH, (long long)LL * LL, HH, 1, 0, 0.125f);
            softmax_kernel<<<dim3(HH * LL), dim3(256), 0, stream>>>(scores, attn_mask, attnh, b);
            // ctx[h] = P_h @ V_h, written interleaved (col stride 64 per head)
            gemm(attnh, vb, nullptr, nullptr, ctxh + (long long)b * LL * DD, LL, DH, LL,
                 LL, DD, DD, (long long)LL * LL, DH, DH, HH, 0, 0, 1.0f);
        }
        // output projection + residual + LN1
        gemm(ctxh, wo_h[l], Lp[l][P_BO], y, nullptr, ML, DD, DD, DD, DD, DD, 0, 0, 0, 1, 0, 0, 1.0f);
        residual_ln_kernel<<<dim3(ML), dim3(256), 0, stream>>>(x, y, Lp[l][P_LN1G], Lp[l][P_LN1B], x, xh);
        // FFN: gelu(x @ w1 + b1) @ w2 + b2, residual + LN2
        gemm(xh, w1_h[l], Lp[l][P_B1], nullptr, hh, ML, FF_, DD, DD, FF_, FF_, 0, 0, 0, 1, 0, 1, 1.0f);
        gemm(hh, w2_h[l], Lp[l][P_B2], y, nullptr, ML, DD, FF_, FF_, DD, DD, 0, 0, 0, 1, 0, 0, 1.0f);
        residual_ln_kernel<<<dim3(ML), dim3(256), 0, stream>>>(x, y, Lp[l][P_LN2G], Lp[l][P_LN2B], x, xh);
    }

    // ---- heads ----
    pool_kernel<<<dim3((BB * DD + 255) / 256), dim3(256), 0, stream>>>(x, pool);
    cls_kernel<<<dim3((BB * LL * NLAB + 255) / 256), dim3(256), 0, stream>>>(x, cls_w, cls_b, tok_log);
    pos_kernel<<<dim3(BB), dim3(1), 0, stream>>>(valid_ids, posbuf);
    init_logits_kernel<<<dim3((BB * LL + 255) / 256), dim3(256), 0, stream>>>(cls_b, logits_v);
    scatter_kernel<<<dim3((BB * LL + 255) / 256), dim3(256), 0, stream>>>(valid_ids, posbuf, tok_log, logits_v);
    logp_kernel<<<dim3((BB * LL + 255) / 256), dim3(256), 0, stream>>>(logits_v, out + 1);
    ce_kernel<<<dim3(1), dim3(256), 0, stream>>>(out + 1, labels, aml, ce_acc);
    rel_kernel<<<dim3(1), dim3(64), 0, stream>>>(pool, rel_w, rel_b, rel_raw);
    final_kernel<<<dim3(1), dim3(1), 0, stream>>>(rel_raw, rel_labels, type_flag, ce_acc, out);
}
